// GIN_66915590472234
// MI455X (gfx1250) — compile-verified, matrix operand order checked by reading, not run on hardware
//
#include <hip/hip_runtime.h>

typedef __attribute__((ext_vector_type(16))) _Float16     v16h;
typedef __attribute__((ext_vector_type(8)))  float        v8f;
typedef __attribute__((ext_vector_type(4)))  unsigned int u32x4;

#define N_NODES 100000
#define N_EDGES 800000

union FragH { v16h v; u32x4 q[2]; };

// A-fragment gather from an LDS row-major f16 tile.
// 16-bit A layout (ISA 7.12.2): lanes 0-15 hold M=lane, K = {kk+0..7, kk+16..23};
// lanes 16-31 hold M=lane-16, K = {kk+8..15, kk+24..31}.
__device__ inline v16h lds_a_frag(const _Float16* rowp, int kk, int lane) {
    const int base = (lane < 16) ? 0 : 8;
    const _Float16* p = rowp + kk + base;           // 16B aligned by construction
    FragH f;
    f.q[0] = *reinterpret_cast<const u32x4*>(p);        // K = kk+base   .. +7
    f.q[1] = *reinterpret_cast<const u32x4*>(p + 16);   // K = kk+base+16.. +23
    return f.v;
}

// B-fragment from pre-packed weights: 32 contiguous bytes per lane.
__device__ inline v16h gmem_b_frag(const _Float16* pack, int kt, int ntiles, int nt, int lane) {
    const _Float16* p = pack + (((size_t)(kt * ntiles + nt) * 32 + lane) << 4);
    FragH f;
    f.q[0] = *reinterpret_cast<const u32x4*>(p);
    f.q[1] = *reinterpret_cast<const u32x4*>(p + 8);
    return f.v;
}

// Pack f32 weight matrix [Kdim, Ndim] (row-major, [in,out]) into per-lane
// WMMA B-fragment order: P[((kt*ntiles+nt)*32+lane)*16 + e]
__global__ void pack_w_kernel(const float* __restrict__ W, _Float16* __restrict__ P,
                              int Kdim, int Ndim) {
    int tid = blockIdx.x * blockDim.x + threadIdx.x;
    int total = Kdim * Ndim;
    if (tid >= total) return;
    int e    = tid & 15;
    int lane = (tid >> 4) & 31;
    int tile = tid >> 9;
    int ntiles = Ndim >> 4;
    int nt = tile % ntiles;
    int kt = tile / ntiles;
    int base = (lane < 16) ? 0 : 8;
    int k = kt * 32 + base + (e & 7) + ((e & 8) ? 16 : 0);
    int n = nt * 16 + (lane & 15);
    P[tid] = (_Float16)W[(size_t)k * Ndim + n];
}

// z = (1 + eps) * x  (also re-initializes scratch every call -> deterministic work)
__global__ void init_scaled_kernel(const float* __restrict__ x, float* __restrict__ z,
                                   const float* __restrict__ eps, int total) {
    int i = blockIdx.x * blockDim.x + threadIdx.x;
    if (i < total) z[i] = (1.0f + eps[0]) * x[i];
}

// segment-sum: z[dst] += feat[src], C channels, one thread per (edge, 4-ch group)
__global__ void scatter_add_kernel(const float* __restrict__ feat, float* __restrict__ z,
                                   const int* __restrict__ edges, int cgroups, int C) {
    int gid = blockIdx.x * blockDim.x + threadIdx.x;
    int e = gid / cgroups;
    if (e >= N_EDGES) return;
    int g = (gid - e * cgroups) * 4;
    int s = edges[e];
    int d = edges[N_EDGES + e];
    const float4 v = *reinterpret_cast<const float4*>(feat + (size_t)s * C + g);
    float* zp = z + (size_t)d * C + g;
    atomicAdd(zp + 0, v.x);
    atomicAdd(zp + 1, v.y);
    atomicAdd(zp + 2, v.z);
    atomicAdd(zp + 3, v.w);
}

// Fused GIN MLP for a tile of 16 nodes:
//   out = [relu_out?relu:id]( relu(zin @ Wa + ba) @ Wb + bb )
// 256 threads = 8 wave32; wave w owns output N-tile w of each GEMM.
template <int CIN, int CHID, int COUT, bool RELU_OUT>
__global__ void __launch_bounds__(256)
gin_mlp_kernel(const float* __restrict__ zin,
               const _Float16* __restrict__ Wa_p, const float* __restrict__ ba,
               const _Float16* __restrict__ Wb_p, const float* __restrict__ bb,
               float* __restrict__ out) {
    constexpr int PAD = 8;  // keeps 16B row alignment, staggers banks
    __shared__ __align__(16) _Float16 sA[16][CIN + PAD];
    __shared__ __align__(16) _Float16 sI[16][CHID + PAD];

    const int m0   = blockIdx.x * 16;
    const int tid  = threadIdx.x;
    const int wave = tid >> 5;
    const int lane = tid & 31;
    const int col  = lane & 15;
    const int rbase = (lane < 16) ? 0 : 8;

    // stage input tile (f32 -> f16) into LDS
    for (int i = tid; i < 16 * CIN; i += 256) {
        int r = i / CIN, c = i - r * CIN;
        sA[r][c] = (_Float16)zin[(size_t)(m0 + r) * CIN + c];
    }
    __syncthreads();

    // GEMM 1: [16,CIN] x [CIN,CHID], wave handles N-tile n0 = wave*16
    {
        const int n0 = wave * 16;
        v8f acc;
        float bv = ba[n0 + col];
        #pragma unroll
        for (int v = 0; v < 8; ++v) acc[v] = bv;
        #pragma unroll
        for (int kt = 0; kt < CIN / 32; ++kt) {
            v16h a = lds_a_frag(&sA[col][0], kt * 32, lane);
            v16h b = gmem_b_frag(Wa_p, kt, CHID / 16, wave, lane);
            acc = __builtin_amdgcn_wmma_f32_16x16x32_f16(
                false, a, false, b, (short)0, acc, false, false);
        }
        // ReLU + spill intermediate to LDS (C/D layout: VGPR v -> M = rbase+v, N = n0+col)
        #pragma unroll
        for (int v = 0; v < 8; ++v) {
            float r = acc[v];
            sI[rbase + v][n0 + col] = (_Float16)(r > 0.0f ? r : 0.0f);
        }
    }
    __syncthreads();

    // GEMM 2: [16,CHID] x [CHID,COUT]
    if (wave < COUT / 16) {
        const int n0 = wave * 16;
        v8f acc;
        float bv = bb[n0 + col];
        #pragma unroll
        for (int v = 0; v < 8; ++v) acc[v] = bv;
        #pragma unroll
        for (int kt = 0; kt < CHID / 32; ++kt) {
            v16h a = lds_a_frag(&sI[col][0], kt * 32, lane);
            v16h b = gmem_b_frag(Wb_p, kt, COUT / 16, wave, lane);
            acc = __builtin_amdgcn_wmma_f32_16x16x32_f16(
                false, a, false, b, (short)0, acc, false, false);
        }
        #pragma unroll
        for (int v = 0; v < 8; ++v) {
            float r = acc[v];
            if (RELU_OUT) r = (r > 0.0f ? r : 0.0f);
            out[(size_t)(m0 + rbase + v) * COUT + n0 + col] = r;
        }
    }
}

extern "C" void kernel_launch(void* const* d_in, const int* in_sizes, int n_in,
                              void* d_out, int out_size, void* d_ws, size_t ws_size,
                              hipStream_t stream) {
    const float* x    = (const float*)d_in[0];
    const int*   eidx = (const int*)d_in[1];   // [2, E] int32 (JAX w/o x64)
    const float* eps1 = (const float*)d_in[2];
    const float* eps2 = (const float*)d_in[3];
    const float* W1a  = (const float*)d_in[4];
    const float* b1a  = (const float*)d_in[5];
    const float* W1b  = (const float*)d_in[6];
    const float* b1b  = (const float*)d_in[7];
    const float* W2a  = (const float*)d_in[8];
    const float* b2a  = (const float*)d_in[9];
    const float* W2b  = (const float*)d_in[10];
    const float* b2b  = (const float*)d_in[11];

    char* ws = (char*)d_ws;
    size_t off = 0;
    auto carve = [&](size_t bytes) {
        void* p = ws + off;
        off = (off + bytes + 255) & ~(size_t)255;
        return p;
    };
    float*    zbuf  = (float*)carve((size_t)N_NODES * 128 * sizeof(float)); // z1 then z2
    float*    hbuf  = (float*)carve((size_t)N_NODES * 128 * sizeof(float)); // layer-1 output
    _Float16* W1a_p = (_Float16*)carve((size_t)64 * 128 * 2);
    _Float16* W1b_p = (_Float16*)carve((size_t)128 * 128 * 2);
    _Float16* W2a_p = (_Float16*)carve((size_t)128 * 128 * 2);
    _Float16* W2b_p = (_Float16*)carve((size_t)128 * 64 * 2);

    // pack weights into per-lane WMMA B-fragment layout (f16)
    pack_w_kernel<<<(64 * 128 + 255) / 256, 256, 0, stream>>>(W1a, W1a_p, 64, 128);
    pack_w_kernel<<<(128 * 128 + 255) / 256, 256, 0, stream>>>(W1b, W1b_p, 128, 128);
    pack_w_kernel<<<(128 * 128 + 255) / 256, 256, 0, stream>>>(W2a, W2a_p, 128, 128);
    pack_w_kernel<<<(128 * 64 + 255) / 256, 256, 0, stream>>>(W2b, W2b_p, 128, 64);

    // ---- layer 1 ----
    init_scaled_kernel<<<(N_NODES * 64 + 255) / 256, 256, 0, stream>>>(x, zbuf, eps1, N_NODES * 64);
    scatter_add_kernel<<<(unsigned)(((size_t)N_EDGES * 16 + 255) / 256), 256, 0, stream>>>(
        x, zbuf, eidx, 16, 64);
    gin_mlp_kernel<64, 128, 128, true><<<N_NODES / 16, 256, 0, stream>>>(
        zbuf, W1a_p, b1a, W1b_p, b1b, hbuf);

    // ---- layer 2 ----
    init_scaled_kernel<<<(N_NODES * 128 + 255) / 256, 256, 0, stream>>>(hbuf, zbuf, eps2, N_NODES * 128);
    scatter_add_kernel<<<(unsigned)(((size_t)N_EDGES * 32 + 255) / 256), 256, 0, stream>>>(
        hbuf, zbuf, eidx, 32, 128);
    gin_mlp_kernel<128, 128, 64, false><<<N_NODES / 16, 256, 0, stream>>>(
        zbuf, W2a_p, b2a, W2b_p, b2b, (float*)d_out);
}
